// Clust_GRU_40690520162859
// MI455X (gfx1250) — matrix operand order repaired and checked by reading, not compiled, not run on hardware
//
#include <hip/hip_runtime.h>
#include <hip/hip_bf16.h>
#include <stdint.h>

typedef __attribute__((ext_vector_type(16))) _Float16 v16h;
typedef __attribute__((ext_vector_type(8)))  float    v8f;

#define SUB_NO 20
#define E_NO   2000
#define I_NO   400
#define H_NO   64
#define G_NO   192   // 3*H
#define BB     8
#define TT     2000
#define BT     16000 // B*T

union AFrag { v16h v; uint32_t u[8]; };

__device__ __forceinline__ float fast_rcp(float x) {
    return __builtin_amdgcn_rcpf(x);   // raw v_rcp_f32
}
__device__ __forceinline__ float fast_sigmoid(float x) {
    return fast_rcp(1.0f + __expf(-x));
}
__device__ __forceinline__ float fast_tanh(float x) {
    return fmaf(2.0f, fast_rcp(1.0f + __expf(-2.0f * x)), -1.0f);
}

// Workgroup barrier that drains ONLY the LDS counter (dscnt), leaving global
// prefetch loads in flight across the barrier. Safe here: all cross-wave
// communication in the GRU loop is through LDS; global traffic is either
// read-only (S_sub prefetch) or per-thread-private (sub_out stores consumed
// by a later kernel launch).
__device__ __forceinline__ void wg_barrier_lds() {
    asm volatile("s_wait_dscnt 0x0\n\t"
                 "s_barrier_signal -1\n\t"
                 "s_barrier_wait -1" ::: "memory");
}

// ---------------------------------------------------------------------------
// Kernel 1: softmax over subunit axis + synaptic scaling.
// ---------------------------------------------------------------------------
__global__ void prep_kernel(const float* __restrict__ Esc, const float* __restrict__ Isc,
                            const float* __restrict__ Ce_raw, const float* __restrict__ Ci_raw,
                            const int* __restrict__ temp,
                            float* __restrict__ Ce_out, float* __restrict__ Ci_out,
                            float* __restrict__ Dte, float* __restrict__ Dti) {
    int c = blockIdx.x * blockDim.x + threadIdx.x;
    float invT = 1.0f / (float)temp[0];
    if (c < E_NO) {
        float v[SUB_NO];
        float m = -3.4e38f;
        #pragma unroll
        for (int s = 0; s < SUB_NO; ++s) { v[s] = Ce_raw[s * E_NO + c] * invT; m = fmaxf(m, v[s]); }
        float sum = 0.f;
        #pragma unroll
        for (int s = 0; s < SUB_NO; ++s) { v[s] = __expf(v[s] - m); sum += v[s]; }
        float inv = fast_rcp(sum);
        float sc  = __expf(Esc[c]);
        #pragma unroll
        for (int s = 0; s < SUB_NO; ++s) {
            float p = v[s] * inv;
            Ce_out[s * E_NO + c] = p;
            Dte[c * SUB_NO + s]  = p * sc;
        }
    } else if (c < E_NO + I_NO) {
        int ci = c - E_NO;
        float v[SUB_NO];
        float m = -3.4e38f;
        #pragma unroll
        for (int s = 0; s < SUB_NO; ++s) { v[s] = Ci_raw[s * I_NO + ci] * invT; m = fmaxf(m, v[s]); }
        float sum = 0.f;
        #pragma unroll
        for (int s = 0; s < SUB_NO; ++s) { v[s] = __expf(v[s] - m); sum += v[s]; }
        float inv = fast_rcp(sum);
        float sc  = -__expf(Isc[ci]);
        #pragma unroll
        for (int s = 0; s < SUB_NO; ++s) {
            float p = v[s] * inv;
            Ci_out[s * I_NO + ci] = p;
            Dti[ci * SUB_NO + s]  = p * sc;
        }
    }
}

// ---------------------------------------------------------------------------
// Kernel 2: S_sub[r][s] = sum_e S_e[r,e]*D_e[e,s] + sum_i S_i[r,i]*D_i[i,s]
// ---------------------------------------------------------------------------
__global__ void __launch_bounds__(256) ssub_kernel(const float* __restrict__ Se,
                                                   const float* __restrict__ Si,
                                                   const float* __restrict__ Dte,
                                                   const float* __restrict__ Dti,
                                                   float* __restrict__ Ssub) {
    __shared__ float dl[100 * SUB_NO];
    int r = blockIdx.x * 256 + threadIdx.x;
    const bool ok = (r < BT);
    float acc[SUB_NO];
    #pragma unroll
    for (int s = 0; s < SUB_NO; ++s) acc[s] = 0.f;

    for (int e0 = 0; e0 < E_NO; e0 += 100) {
        __syncthreads();
        for (int idx = threadIdx.x; idx < 100 * SUB_NO; idx += 256) dl[idx] = Dte[e0 * SUB_NO + idx];
        __syncthreads();
        if (ok) {
            for (int ee = 0; ee < 100; ++ee) {
                float x = Se[r * E_NO + e0 + ee];
                const float* dp = &dl[ee * SUB_NO];
                #pragma unroll
                for (int s = 0; s < SUB_NO; ++s) acc[s] = fmaf(x, dp[s], acc[s]);
            }
        }
    }
    for (int i0 = 0; i0 < I_NO; i0 += 100) {
        __syncthreads();
        for (int idx = threadIdx.x; idx < 100 * SUB_NO; idx += 256) dl[idx] = Dti[i0 * SUB_NO + idx];
        __syncthreads();
        if (ok) {
            for (int ii = 0; ii < 100; ++ii) {
                float x = Si[r * I_NO + i0 + ii];
                const float* dp = &dl[ii * SUB_NO];
                #pragma unroll
                for (int s = 0; s < SUB_NO; ++s) acc[s] = fmaf(x, dp[s], acc[s]);
            }
        }
    }
    if (ok) {
        #pragma unroll
        for (int s = 0; s < SUB_NO; ++s) Ssub[r * SUB_NO + s] = acc[s];
    }
}

// ---------------------------------------------------------------------------
// Kernel 3: sequential GRU per subunit. 1 block / subunit, 128 threads (4 waves).
// Wave w owns h-slice [w*16, w*16+16) of ALL 3 gates: gates stay in registers,
// one LDS-only barrier per step, W_hh pinned in VGPRs as f16 B-fragments.
// ---------------------------------------------------------------------------
__global__ void __launch_bounds__(128) gru_kernel(const float* __restrict__ W_ih,
                                                  const float* __restrict__ W_hh,
                                                  const float* __restrict__ b_ih,
                                                  const float* __restrict__ b_hh,
                                                  const float* __restrict__ W_lin,
                                                  const float* __restrict__ b_lin,
                                                  const float* __restrict__ Ssub,
                                                  float* __restrict__ sub_out) {
    const int s    = blockIdx.x;
    const int tid  = threadIdx.x;
    const int lane = tid & 31;
    const int w    = tid >> 5;
    const int half = lane >> 4;       // 0: b=0..3, 1: b=4..7
    const int lcol = lane & 15;
    const int hcol = w * 16 + lcol;   // this lane's h column

    __shared__ __align__(16) _Float16 hbuf[2][16 * H_NO];  // f16 state, parity buffered
    __shared__ __align__(16) float    ssub_s[2][BB];
    __shared__ __align__(16) float    wsum[2][4 * BB];     // per-wave readout partials

    // loop-invariant per-lane gate parameters (h = hcol)
    const float wih_r = W_ih[s * G_NO + hcol];
    const float wih_z = W_ih[s * G_NO + H_NO + hcol];
    const float wih_n = W_ih[s * G_NO + 2 * H_NO + hcol];
    const float bih_r = b_ih[s * G_NO + hcol];
    const float bih_z = b_ih[s * G_NO + H_NO + hcol];
    const float bih_n = b_ih[s * G_NO + 2 * H_NO + hcol];
    const float bhh_r = b_hh[s * G_NO + hcol];
    const float bhh_z = b_hh[s * G_NO + H_NO + hcol];
    const float bhh_n = b_hh[s * G_NO + 2 * H_NO + hcol];
    const float wlin_h = W_lin[s * H_NO + hcol];
    const float blin   = b_lin[s];

    // B fragments in VGPRs: tile j covers gate-cols g = j*64 + w*16 + [0,16)
    v16h bf[3][2];
    const float* Whh_s = W_hh + s * G_NO * H_NO;
    #pragma unroll
    for (int j = 0; j < 3; ++j) {
        int g = j * H_NO + w * 16 + lcol;
        #pragma unroll
        for (int kc = 0; kc < 2; ++kc) {
            #pragma unroll
            for (int v = 0; v < 8; ++v) {
                int k = kc * 32 + (half << 4) + 2 * v;
                bf[j][kc][2 * v]     = (_Float16)Whh_s[g * H_NO + k];
                bf[j][kc][2 * v + 1] = (_Float16)Whh_s[g * H_NO + k + 1];
            }
        }
    }
    // Pin the converted f16 fragments (prevent per-iteration rematerialization).
    #pragma unroll
    for (int j = 0; j < 3; ++j) {
        asm volatile("" : "+v"(bf[j][0]));
        asm volatile("" : "+v"(bf[j][1]));
    }

    for (int i = tid; i < 2 * 16 * H_NO; i += 128) ((_Float16*)hbuf)[i] = (_Float16)0.0f;

    float hF[4] = {0.f, 0.f, 0.f, 0.f};   // f32 state for (b = half*4+q, h = hcol)

    // publish t=0 drive synchronously
    if (tid < BB) ssub_s[0][tid] = Ssub[(tid * TT + 0) * SUB_NO + s];
    __syncthreads();

    float nexts = 0.f;
    for (int t = 0; t < TT; ++t) {
        const int p = t & 1;
        const uint32_t* hsrc = (const uint32_t*)hbuf[1 - p];  // state t-1

        // ---- issue prefetch of next step's drive EARLY (hidden by this step) ----
        if (tid < BB && t + 1 < TT) nexts = Ssub[(tid * TT + t + 1) * SUB_NO + s];

        // ---- A fragments (16x32 f16) + WMMA, gates land in registers ----
        AFrag a0, a1;
        {
            int m  = lcol;
            int hk = half * 8;
            #pragma unroll
            for (int v = 0; v < 8; ++v) {
                int k0 = hk * 2 + v * 2;
                a0.u[v] = hsrc[m * 32 + (k0 >> 1)];
                a1.u[v] = hsrc[m * 32 + ((k0 + 32) >> 1)];
            }
        }
        v8f acc[3];
        #pragma unroll
        for (int j = 0; j < 3; ++j) {
            v8f c = {};
            c = __builtin_amdgcn_wmma_f32_16x16x32_f16(false, a0.v, false, bf[j][0],
                                                       (short)0, c, false, false);
            c = __builtin_amdgcn_wmma_f32_16x16x32_f16(false, a1.v, false, bf[j][1],
                                                       (short)0, c, false, false);
            acc[j] = c;
        }

        // ---- drain previous step's readout while WMMA retires ----
        if (t > 0 && tid < BB) {
            const int pp = (t - 1) & 1;
            float sum = wsum[pp][tid] + wsum[pp][BB + tid] +
                        wsum[pp][2 * BB + tid] + wsum[pp][3 * BB + tid];
            sub_out[(tid * TT + (t - 1)) * SUB_NO + s] = sum + blin;
        }

        // ---- register-resident GRU nonlinearity: 4 cells (b=half*4+q, h=hcol) ----
        {
            const float* xsp = &ssub_s[p][half * 4];
            _Float16* hdst = hbuf[p];
            float psum[4];
            #pragma unroll
            for (int q = 0; q < 4; ++q) {
                float rr_raw = half ? acc[0][4 + q] : acc[0][q];
                float zz_raw = half ? acc[1][4 + q] : acc[1][q];
                float nn_raw = half ? acc[2][4 + q] : acc[2][q];
                float x  = xsp[q];
                float r  = fast_sigmoid(fmaf(x, wih_r, bih_r) + rr_raw + bhh_r);
                float z  = fast_sigmoid(fmaf(x, wih_z, bih_z) + zz_raw + bhh_z);
                float nn = fast_tanh(fmaf(x, wih_n, bih_n) + r * (nn_raw + bhh_n));
                float hn = (1.0f - z) * nn + z * hF[q];
                hF[q] = hn;
                _Float16 hh = (_Float16)hn;
                int b = half * 4 + q;
                hdst[b * H_NO + hcol]       = hh;   // batch row
                hdst[(b + 8) * H_NO + hcol] = hh;   // duplicate into padding row
                psum[q] = hn * wlin_h;
            }
            // 16-lane reduction per half (all lanes in a 16-group share b range)
            #pragma unroll
            for (int off = 8; off >= 1; off >>= 1) {
                #pragma unroll
                for (int q = 0; q < 4; ++q) psum[q] += __shfl_xor(psum[q], off, 32);
            }
            if (lcol == 0) {
                float4 st = make_float4(psum[0], psum[1], psum[2], psum[3]);
                *(float4*)&wsum[p][w * BB + half * 4] = st;
            }
        }

        // ---- publish drive for t+1 (into other parity buf) ----
        if (tid < BB) ssub_s[1 - p][tid] = nexts;

        wg_barrier_lds();   // single LDS-only barrier per step
    }

    // final step's readout
    if (tid < BB) {
        const int pp = (TT - 1) & 1;
        float sum = wsum[pp][tid] + wsum[pp][BB + tid] +
                    wsum[pp][2 * BB + tid] + wsum[pp][3 * BB + tid];
        sub_out[(tid * TT + (TT - 1)) * SUB_NO + s] = sum + blin;
    }
}

// ---------------------------------------------------------------------------
// Kernel 4: final[b,t] = sum_s sub_out[b,t,s] + V_o
// ---------------------------------------------------------------------------
__global__ void final_kernel(const float* __restrict__ sub_out,
                             const float* __restrict__ V_o,
                             float* __restrict__ fin) {
    int r = blockIdx.x * 256 + threadIdx.x;
    if (r < BT) {
        float s = V_o[0];
        #pragma unroll
        for (int k = 0; k < SUB_NO; ++k) s += sub_out[r * SUB_NO + k];
        fin[r] = s;
    }
}

extern "C" void kernel_launch(void* const* d_in, const int* in_sizes, int n_in,
                              void* d_out, int out_size, void* d_ws, size_t ws_size,
                              hipStream_t stream) {
    const float* S_e     = (const float*)d_in[0];
    const float* S_i     = (const float*)d_in[1];
    const float* E_scale = (const float*)d_in[2];
    const float* I_scale = (const float*)d_in[3];
    const float* Ce_raw  = (const float*)d_in[4];
    const float* Ci_raw  = (const float*)d_in[5];
    const float* W_ih    = (const float*)d_in[6];
    const float* W_hh    = (const float*)d_in[7];
    const float* b_ih    = (const float*)d_in[8];
    const float* b_hh    = (const float*)d_in[9];
    const float* W_lin   = (const float*)d_in[10];
    const float* b_lin   = (const float*)d_in[11];
    const float* V_o     = (const float*)d_in[12];
    const int*   temp    = (const int*)d_in[13];

    float* out     = (float*)d_out;
    float* fin     = out;                      // [B,T]        16000
    float* sub_out = out + BT;                 // [B,T,S]     320000
    float* Ce_out  = out + BT + BT * SUB_NO;   // [S,E]        40000
    float* Ci_out  = Ce_out + SUB_NO * E_NO;   // [S,I]         8000

    float* ws   = (float*)d_ws;
    float* Dte  = ws;                                  // [E][S]  40000
    float* Dti  = ws + E_NO * SUB_NO;                  // [I][S]   8000
    float* Ssub = ws + (E_NO + I_NO) * SUB_NO;         // [B,T,S] 320000

    prep_kernel<<<(E_NO + I_NO + 255) / 256, 256, 0, stream>>>(
        E_scale, I_scale, Ce_raw, Ci_raw, temp, Ce_out, Ci_out, Dte, Dti);

    ssub_kernel<<<(BT + 255) / 256, 256, 0, stream>>>(S_e, S_i, Dte, Dti, Ssub);

    gru_kernel<<<SUB_NO, 128, 0, stream>>>(W_ih, W_hh, b_ih, b_hh, W_lin, b_lin,
                                           Ssub, sub_out);

    final_kernel<<<(BT + 255) / 256, 256, 0, stream>>>(sub_out, V_o, fin);
}
